// ConformerMultiHeadedSelfAttention_55336358641788
// MI455X (gfx1250) — matrix operand order repaired and checked by reading, not compiled, not run on hardware
//
#include <hip/hip_runtime.h>
#include <hip/hip_bf16.h>

typedef __attribute__((ext_vector_type(16))) __bf16 v16bf;
typedef __attribute__((ext_vector_type(8)))  float  v8f;

#define B_  4
#define S_  1024
#define D_  512
#define H_  16
#define DH_ 32

__device__ __forceinline__ unsigned short f32_bf16(float f) {
  unsigned int u = __float_as_uint(f);
  unsigned int r = u + 0x7FFFu + ((u >> 16) & 1u);   // round-to-nearest-even
  return (unsigned short)(r >> 16);
}
__device__ __forceinline__ float bf16_f32(unsigned short h) {
  return __uint_as_float(((unsigned int)h) << 16);
}

// LDS byte offset of a generic pointer to __shared__ (ISA: LDS flat addr low 32
// bits are the LDS offset).
__device__ __forceinline__ unsigned lds_off(const void* p) {
  return (unsigned)(unsigned long long)(uintptr_t)p;
}

// ---------------------------------------------------------------- convert
__global__ void cvt_f32_bf16_kernel(const float* __restrict__ src,
                                    unsigned short* __restrict__ dst, int n) {
  int i = blockIdx.x * blockDim.x + threadIdx.x;
  if (i < n) dst[i] = f32_bf16(src[i]);
}

// bqu[d] = bq[d] + u[d], bqv[d] = bq[d] + v[d]  (u_bias/v_bias flatten to d)
__global__ void bias_combine_kernel(const float* __restrict__ bq,
                                    const float* __restrict__ u,
                                    const float* __restrict__ v,
                                    float* __restrict__ bqu,
                                    float* __restrict__ bqv) {
  int d = blockIdx.x * blockDim.x + threadIdx.x;
  if (d < D_) { bqu[d] = bq[d] + u[d]; bqv[d] = bq[d] + v[d]; }
}

// ---------------------------------------------------------------- WMMA GEMM
// Compile-time specialized; block = 256 threads = 8 waves; block tile 64 x BN.
// LDS always holds B as [n][k] so the B fragment gather is contiguous pairs
// (merges into ds_load_b128). Staging:
//   A (bf16 source):  async global->LDS b128 (ASYNCcnt, double-buffered)
//   A (f32 source):   synchronous convert while staging (ctx GEMM)
//   B (B_TRANS, [NxK] in memory): async global->LDS b128/b64
//   B (!B_TRANS, [KxN] in memory): sync b128 read + ds_store_b16 transpose
// Tile i+1 is in flight while WMMA consumes tile i. Each wave owns one 16-row
// m-tile and TPW 16-col n-tiles, reusing its A fragment across TPW wmmas.
// All dims are multiples of the tile sizes -> no bounds checks; EXEC stays
// all-ones around v_wmma (ISA requirement).
template<int BN, bool A_F32, bool B_TRANS, bool C_BF16, bool HAS_BIAS>
__global__ __launch_bounds__(256)
void gemm_wmma(const void* __restrict__ Ap, long long sAb, long long sAh, int lda,
               const unsigned short* __restrict__ Bp, long long sBb, long long sBh, int ldb,
               void* __restrict__ Cp, long long sCb, long long sCh, int ldc,
               const float* __restrict__ bias, int K, int zH) {
  constexpr int BM  = 64;
  constexpr int NT  = BN / 16;               // n-tiles per block
  constexpr int TPW = (BM / 16) * NT / 8;    // tiles per wave (2 for BN=64, 1 for BN=32)
  constexpr int EPT = BN / 8;                // B elems per thread (8 -> b128, 4 -> b64)
  constexpr int TPR = B_TRANS ? (32 / EPT) : 8;   // threads per B source row
  constexpr int LPS = (A_F32 ? 0 : 1) + (B_TRANS ? 1 : 0);  // async loads/stage/thread

  __shared__ __align__(16) unsigned short As[2][BM][40];   // [m][k], 80B rows
  __shared__ __align__(16) unsigned short Bs[2][BN][40];   // [n][k], 80B rows

  const int zb = blockIdx.z / zH;
  const int zh = blockIdx.z % zH;
  const long long offA = (long long)zb * sAb + (long long)zh * sAh;
  const long long offB = (long long)zb * sBb + (long long)zh * sBh;
  const long long offC = (long long)zb * sCb + (long long)zh * sCh;

  const int n0 = blockIdx.x * BN;
  const int m0 = blockIdx.y * BM;

  const int t    = threadIdx.x;
  const int lane = t & 31;
  const int wave = t >> 5;                   // 0..7
  const int mt   = (wave * TPW) / NT;        // m-tile of this wave
  const int ln15 = lane & 15;

  // loader coords: A = BM x 32, 8 contiguous bf16 (16B) per thread
  const int lrA = t >> 2;                    // 0..63
  const int lcA = (t & 3) * 8;
  const int lrB = t / TPR;                   // NT: n row; NN: k row
  const int lcB = (t % TPR) * EPT;           // NT: k col; NN: n col

  const unsigned short* Abf  = (const unsigned short*)Ap;
  const float*          Af32 = (const float*)Ap;

  // stage one 32-deep K slab into LDS buffer `buf`
  auto stage = [&](int kt, int buf) {
    const long long baseA = offA + (long long)(m0 + lrA) * lda + (kt + lcA);
    if constexpr (A_F32) {
      #pragma unroll
      for (int j = 0; j < 8; ++j) As[buf][lrA][lcA + j] = f32_bf16(Af32[baseA + j]);
    } else {
      unsigned la = lds_off(&As[buf][lrA][lcA]);
      asm volatile("global_load_async_to_lds_b128 %0, %1, off"
                   :: "v"(la), "v"(Abf + baseA) : "memory");
    }
    if constexpr (B_TRANS) {   // memory [N x K]: rows already k-contiguous
      const long long baseB = offB + (long long)(n0 + lrB) * ldb + (kt + lcB);
      unsigned lb = lds_off(&Bs[buf][lrB][lcB]);
      if constexpr (EPT == 8) {
        asm volatile("global_load_async_to_lds_b128 %0, %1, off"
                     :: "v"(lb), "v"(Bp + baseB) : "memory");
      } else {
        asm volatile("global_load_async_to_lds_b64 %0, %1, off"
                     :: "v"(lb), "v"(Bp + baseB) : "memory");
      }
    } else {                   // memory [K x N]: read row k, scatter to Bs[n][k]
      const long long baseB = offB + (long long)(kt + lrB) * ldb + (n0 + lcB);
      unsigned short tmp[EPT];
      #pragma unroll
      for (int j = 0; j < EPT; ++j) tmp[j] = Bp[baseB + j];
      #pragma unroll
      for (int j = 0; j < EPT; ++j) Bs[buf][lcB + j][lrB] = tmp[j];
    }
  };

  v8f acc[TPW];
  #pragma unroll
  for (int j = 0; j < TPW; ++j) acc[j] = (v8f){0.f,0.f,0.f,0.f,0.f,0.f,0.f,0.f};

  stage(0, 0);
  if (K > 32) stage(32, 1);

  for (int kt = 0; kt < K; kt += 32) {
    const int buf = (kt >> 5) & 1;

    // wait until this buffer's async copies land (next tile may stay in flight)
    if constexpr (LPS > 0) {
      if (kt + 32 < K) {
        if constexpr (LPS == 2) asm volatile("s_wait_asynccnt 2" ::: "memory");
        else                    asm volatile("s_wait_asynccnt 1" ::: "memory");
      } else {
        asm volatile("s_wait_asynccnt 0" ::: "memory");
      }
    }
    __syncthreads();

    // --- A fragment (ISA 16-bit A 16x32 layout), shared across TPW wmmas
    union { unsigned int u[8]; v16bf v; } afrag;
    const int mrow = mt * 16 + ln15;
    const int kbA  = (lane >> 4) * 8;   // lane halves: K {0..7,16..23} / {8..15,24..31}
    #pragma unroll
    for (int i = 0; i < 8; ++i) {
      const int ka = kbA + 2 * i + ((i >= 4) ? 8 : 0);
      afrag.u[i] = *(const unsigned int*)&As[buf][mrow][ka];
    }

    // --- B fragments (contiguous pairs from Bs[n][k]) + wmma per owned n-tile
    #pragma unroll
    for (int j = 0; j < TPW; ++j) {
      const int nt = (wave * TPW + j) % NT;
      union { unsigned int u[8]; v16bf v; } bfrag;
      const int kbB = (lane >> 4) * 8;   // lane halves: K {0..7,16..23} / {8..15,24..31}
      #pragma unroll
      for (int i = 0; i < 8; ++i) {
        bfrag.u[i] = *(const unsigned int*)&Bs[buf][nt * 16 + ln15][kbB + 2 * i + ((i >= 4) ? 8 : 0)];
      }
      acc[j] = __builtin_amdgcn_wmma_f32_16x16x32_bf16(
          false, afrag.v, false, bfrag.v, (short)0, acc[j], false, false);
    }
    __syncthreads();   // all waves done reading buf before it is refilled

    if (kt + 64 < K) stage(kt + 64, buf);
  }

  // --- writeback (f32 C/D layout: VGPR r -> rows r / r+8 by lane half)
  float*          Cf = (float*)Cp;
  unsigned short* Cb = (unsigned short*)Cp;
  #pragma unroll
  for (int j = 0; j < TPW; ++j) {
    const int nt  = (wave * TPW + j) % NT;
    const int col = n0 + nt * 16 + ln15;
    const float bv = HAS_BIAS ? bias[col] : 0.f;
    #pragma unroll
    for (int r = 0; r < 8; ++r) {
      const int row = m0 + mt * 16 + ((lane < 16) ? r : (r + 8));
      const float val = acc[j][r] + bv;
      const long long idx = offC + (long long)row * ldc + col;
      if (C_BF16) Cb[idx] = f32_bf16(val);
      else        Cf[idx] = val;
    }
  }
}

// ------------------------------------------------- fused rel-shift + softmax
// attn region of d_out holds content scores (f32); posraw is bf16 [B,H,S,S].
// score[q,k] = (content + shifted_pos) * scale; mask keys; softmax; in-place.
__global__ __launch_bounds__(256)
void softmax_relshift_kernel(float* __restrict__ attn,
                             const unsigned short* __restrict__ posraw,
                             const int* __restrict__ mask,
                             float scale) {
  const int q = blockIdx.x;
  const int h = blockIdx.y;
  const int b = blockIdx.z;
  const long long bh  = (long long)(b * H_ + h);
  float* crow = attn + (bh * S_ + q) * S_;
  const unsigned short* prow0 = posraw + bh * S_ * S_ + (long long)q * S_;
  const unsigned short* prow1 = prow0 + S_;   // row q+1 (only read when k>q+1)
  const int* mrow = mask + b * S_;

  __shared__ float red[256];
  const int t = threadIdx.x;

  float vals[S_ / 256];
  float vmax = -3.0e38f;
  #pragma unroll
  for (int j = 0; j < S_ / 256; ++j) {
    const int k = t + j * 256;
    float pos;
    if (k <= q)          pos = bf16_f32(prow0[S_ - 1 - q + k]);
    else if (k == q + 1) pos = 0.f;
    else                 pos = bf16_f32(prow1[k - q - 2]);
    float s = (crow[k] + pos) * scale;
    if (mrow[k] != 0) s = -1.0e9f;
    vals[j] = s;
    vmax = fmaxf(vmax, s);
  }
  red[t] = vmax; __syncthreads();
  for (int off = 128; off > 0; off >>= 1) {
    if (t < off) red[t] = fmaxf(red[t], red[t + off]);
    __syncthreads();
  }
  vmax = red[0]; __syncthreads();

  float sum = 0.f;
  #pragma unroll
  for (int j = 0; j < S_ / 256; ++j) { vals[j] = __expf(vals[j] - vmax); sum += vals[j]; }
  red[t] = sum; __syncthreads();
  for (int off = 128; off > 0; off >>= 1) {
    if (t < off) red[t] += red[t + off];
    __syncthreads();
  }
  const float inv = 1.0f / red[0];
  #pragma unroll
  for (int j = 0; j < S_ / 256; ++j) crow[t + j * 256] = vals[j] * inv;
}

// ---------------------------------------------------------------- launch
extern "C" void kernel_launch(void* const* d_in, const int* in_sizes, int n_in,
                              void* d_out, int out_size, void* d_ws, size_t ws_size,
                              hipStream_t stream) {
  const float* query = (const float*)d_in[0];
  const float* key   = (const float*)d_in[1];
  const float* value = (const float*)d_in[2];
  const int*   mask  = (const int*)d_in[3];
  const float* enc   = (const float*)d_in[4];
  const float* Wq    = (const float*)d_in[5];
  const float* bq    = (const float*)d_in[6];
  const float* Wk    = (const float*)d_in[7];
  const float* Wv    = (const float*)d_in[8];
  const float* Wp    = (const float*)d_in[9];
  const float* ub    = (const float*)d_in[10];
  const float* vb    = (const float*)d_in[11];
  const float* Wo    = (const float*)d_in[12];
  const float* bo    = (const float*)d_in[13];

  const int MQ = B_ * S_;                 // 4096
  char* ws = (char*)d_ws;
  size_t off = 0;
  auto alloc = [&](size_t bytes) -> void* {
    void* p = ws + off; off += (bytes + 255) & ~(size_t)255; return p;
  };

  unsigned short* qin  = (unsigned short*)alloc((size_t)MQ * D_ * 2);
  unsigned short* kin  = (unsigned short*)alloc((size_t)MQ * D_ * 2);
  unsigned short* vin  = (unsigned short*)alloc((size_t)MQ * D_ * 2);
  unsigned short* ein  = (unsigned short*)alloc((size_t)S_ * D_ * 2);
  unsigned short* Wqb  = (unsigned short*)alloc((size_t)D_ * D_ * 2);
  unsigned short* Wkb  = (unsigned short*)alloc((size_t)D_ * D_ * 2);
  unsigned short* Wvb  = (unsigned short*)alloc((size_t)D_ * D_ * 2);
  unsigned short* Wpb  = (unsigned short*)alloc((size_t)D_ * D_ * 2);
  unsigned short* Wob  = (unsigned short*)alloc((size_t)D_ * D_ * 2);
  float*          bqu  = (float*)alloc((size_t)D_ * 4);
  float*          bqv  = (float*)alloc((size_t)D_ * 4);
  unsigned short* qub  = (unsigned short*)alloc((size_t)MQ * D_ * 2);
  unsigned short* qvb  = (unsigned short*)alloc((size_t)MQ * D_ * 2);
  unsigned short* kbf  = (unsigned short*)alloc((size_t)MQ * D_ * 2);
  unsigned short* vbf  = (unsigned short*)alloc((size_t)MQ * D_ * 2);
  unsigned short* pbf  = (unsigned short*)alloc((size_t)S_ * D_ * 2);
  unsigned short* posb = (unsigned short*)alloc((size_t)B_ * H_ * S_ * S_ * 2);
  unsigned short* ctxb = (unsigned short*)alloc((size_t)MQ * D_ * 2);

  float* outf  = (float*)d_out;                       // [B,S,D]
  float* attnf = outf + (size_t)B_ * S_ * D_;         // [B,H,S,S]

  auto cvt = [&](const float* s, unsigned short* d, int n) {
    cvt_f32_bf16_kernel<<<dim3((n + 255) / 256), dim3(256), 0, stream>>>(s, d, n);
  };
  cvt(query, qin, MQ * D_);
  cvt(key,   kin, MQ * D_);
  cvt(value, vin, MQ * D_);
  cvt(enc,   ein, S_ * D_);
  cvt(Wq, Wqb, D_ * D_);
  cvt(Wk, Wkb, D_ * D_);
  cvt(Wv, Wvb, D_ * D_);
  cvt(Wp, Wpb, D_ * D_);
  cvt(Wo, Wob, D_ * D_);

  bias_combine_kernel<<<dim3(2), dim3(256), 0, stream>>>(bq, ub, vb, bqu, bqv);

  const long long SD  = (long long)S_ * D_;
  const long long SS  = (long long)S_ * S_;
  const long long HSS = (long long)H_ * S_ * S_;

  // q+u and q+v projections (head bias folded into column bias), bf16 out
  gemm_wmma<64,false,false,true,true><<<dim3(D_/64, MQ/64, 1), dim3(256), 0, stream>>>(
      qin, 0, 0, D_,  Wqb, 0, 0, D_,  qub, 0, 0, D_,  bqu, D_, 1);
  gemm_wmma<64,false,false,true,true><<<dim3(D_/64, MQ/64, 1), dim3(256), 0, stream>>>(
      qin, 0, 0, D_,  Wqb, 0, 0, D_,  qvb, 0, 0, D_,  bqv, D_, 1);
  // k, v, p projections
  gemm_wmma<64,false,false,true,false><<<dim3(D_/64, MQ/64, 1), dim3(256), 0, stream>>>(
      kin, 0, 0, D_,  Wkb, 0, 0, D_,  kbf, 0, 0, D_,  nullptr, D_, 1);
  gemm_wmma<64,false,false,true,false><<<dim3(D_/64, MQ/64, 1), dim3(256), 0, stream>>>(
      vin, 0, 0, D_,  Wvb, 0, 0, D_,  vbf, 0, 0, D_,  nullptr, D_, 1);
  gemm_wmma<64,false,false,true,false><<<dim3(D_/64, S_/64, 1), dim3(256), 0, stream>>>(
      ein, 0, 0, D_,  Wpb, 0, 0, D_,  pbf, 0, 0, D_,  nullptr, D_, 1);

  // content scores -> attn region of d_out (f32), per (b,h), NT, K=DH
  gemm_wmma<64,false,true,false,false><<<dim3(S_/64, S_/64, B_*H_), dim3(256), 0, stream>>>(
      qub, SD, DH_, D_,  kbf, SD, DH_, D_,
      attnf, HSS, SS, S_,  nullptr, DH_, H_);
  // positional raw scores -> posb (bf16), per (b,h); p has no batch stride
  gemm_wmma<64,false,true,true,false><<<dim3(S_/64, S_/64, B_*H_), dim3(256), 0, stream>>>(
      qvb, SD, DH_, D_,  pbf, 0, DH_, D_,
      posb, HSS, SS, S_,  nullptr, DH_, H_);

  // fused rel_shift + scale + mask + softmax (in-place on attn)
  const float scale = 0.044194173824159216f;   // 1/sqrt(512)
  softmax_relshift_kernel<<<dim3(S_, H_, B_), dim3(256), 0, stream>>>(
      attnf, posb, mask, scale);

  // ctx = attn @ v  (A f32 converted inline, bf16 out), per (b,h)
  gemm_wmma<32,true,false,true,false><<<dim3(1, S_/64, B_*H_), dim3(256), 0, stream>>>(
      attnf, HSS, SS, S_,  vbf, SD, DH_, D_,
      ctxb, SD, DH_, D_,  nullptr, S_, H_);

  // out = ctx @ Wo + bo  (f32 out -> d_out)
  gemm_wmma<64,false,false,false,true><<<dim3(D_/64, MQ/64, 1), dim3(256), 0, stream>>>(
      ctxb, 0, 0, D_,  Wob, 0, 0, D_,
      outf, 0, 0, D_,  bo, D_, 1);
}
